// CustomConv2d_85899346070
// MI455X (gfx1250) — matrix operand order repaired and checked by reading, not compiled, not run on hardware
//
#include <hip/hip_runtime.h>

// CDNA5 / gfx1250: wave32, WMMA 16x16x32 f16 -> f32
typedef __attribute__((ext_vector_type(16))) _Float16 v16h;
typedef __attribute__((ext_vector_type(8)))  float    v8f;

#define IN_W     2048
#define IN_H     2048
#define OUT_W    2033
#define OUT_H    2033
#define KS       16
#define NCH      4
#define TILE_X   16
#define STRIP_H  64
#define STRIPS_X 128   // ceil(2033/16)
#define STRIPS_Y 32    // ceil(2033/64)
#define NWAVES   8
#define BLOCK    (NWAVES*32)

union H2   { unsigned u; _Float16 h[2]; };
union V16U { v16h v; _Float16 h[16]; unsigned u[8]; };

__global__ __launch_bounds__(BLOCK) void psnr_conv_wmma(
    const float* __restrict__ x, const float* __restrict__ ker,
    float* __restrict__ out)
{
  // Per-wave private LDS regions: no inter-wave sync needed anywhere.
  __shared__ unsigned stageS[NWAVES][32];           // packed (x^2, x) f16 pairs, 31 cols
  __shared__ float    accumS[NWAVES][16][NCH][16];  // ring: [slot][ch][xo]
  __shared__ float    skkS[NWAVES][NCH];

  const int lane = threadIdx.x & 31;
  const int wv   = threadIdx.x >> 5;
  const int half = lane >> 4;       // lane half selects K group / M offset
  const int nI   = lane & 15;       // WMMA N index == kernel-row i (for B and D)

  // ---- Build per-channel B matrices in registers + Skk = sum w*k^2 ----
  // B[k=2j,   n=i] = w[i][j]            (multiplies x^2)
  // B[k=2j+1, n=i] = -2*w[i][j]*k_c[i][j] (multiplies x)
  // B VGPR p holds K = 2j + {0,1} with j = (p&3) + 4*half + 8*(p>>2), column n = lane%16.
  V16U Bm[NCH];
  float skkp[NCH];
  #pragma unroll
  for (int c = 0; c < NCH; ++c) skkp[c] = 0.0f;
  #pragma unroll
  for (int p = 0; p < 8; ++p) {
    const int j  = (p & 3) + 4*half + 8*(p >> 2);
    const float k3 = ker[3*KS*KS + nI*KS + j];
    const float a  = k3 * (1.0f/255.0f);
    const float w  = (1.0f - a) * (1.0f - a);
    #pragma unroll
    for (int c = 0; c < NCH; ++c) {
      const float kc = ker[c*KS*KS + nI*KS + j];
      Bm[c].h[2*p]   = (_Float16)w;
      Bm[c].h[2*p+1] = (_Float16)(-2.0f * w * kc);
      skkp[c] += w * kc * kc;
    }
  }
  if (lane < NCH) skkS[wv][lane] = 0.0f;    // same-wave DS ops are in-order
  #pragma unroll
  for (int c = 0; c < NCH; ++c) atomicAdd(&skkS[wv][c], skkp[c]);
  float skk[NCH];
  #pragma unroll
  for (int c = 0; c < NCH; ++c) skk[c] = skkS[wv][c];

  // ---- Zero the ring accumulator ----
  {
    float* a0 = &accumS[wv][0][0][0];
    for (int t = lane; t < 16*NCH*16; t += 32) a0[t] = 0.0f;
  }

  const int strip = blockIdx.x * NWAVES + wv;
  const int sx = strip % STRIPS_X;
  const int sy = strip / STRIPS_X;
  if (sy >= STRIPS_Y) return;
  const int x0   = sx * TILE_X;
  const int y0   = sy * STRIP_H;
  const int yEnd = (y0 + STRIP_H < OUT_H) ? (y0 + STRIP_H) : OUT_H;

  int col = x0 + lane; if (col > IN_W-1) col = IN_W-1;   // clamp right edge
  const int e0 = (lane & 15) + 4*half;                    // Toeplitz gather base

  for (int r = y0; r < yEnd + (KS-1); ++r) {
    #pragma unroll
    for (int c = 0; c < NCH; ++c) {
      // Stage one input-row segment: packed (x^2, x) halves per column.
      if (lane < 31) {
        const size_t base = (size_t)c*(IN_W*IN_H) + (size_t)r*IN_W + (size_t)col;
        const float xv = x[base];
        if (r + 1 < IN_H) __builtin_prefetch(&x[base + IN_W], 0, 1);
        H2 t; t.h[0] = (_Float16)(xv*xv); t.h[1] = (_Float16)xv;
        stageS[wv][lane] = t.u;
      }
      // Gather A (Toeplitz): lane = output col m; VGPR q <-> tap j = 4*half+q,
      // VGPR 4+q <-> tap j = 4*half+8+q  (matches 16-bit A K-layout).
      V16U A;
      #pragma unroll
      for (int q = 0; q < 4; ++q) {
        A.u[q]     = stageS[wv][e0 + q];
        A.u[4 + q] = stageS[wv][e0 + 8 + q];
      }
      v8f cz = {};
      v8f D = __builtin_amdgcn_wmma_f32_16x16x32_f16(
          false, A.v, false, Bm[c].v, (short)0, cz, false, false);
      // Scatter-accumulate: column n=i contributes to output row (r - i);
      // lane owns (slot, c, xo-range) exclusively -> plain RMW, no atomics.
      const int orow = r - nI;
      if (orow >= y0 && orow < yEnd) {
        const int slot = orow & 15;
        float4* dst = (float4*)&accumS[wv][slot][c][half*8];  // 16B aligned
        float4 s0 = dst[0], s1 = dst[1];
        s0.x += D[0]; s0.y += D[1]; s0.z += D[2]; s0.w += D[3];
        s1.x += D[4]; s1.y += D[5]; s1.z += D[6]; s1.w += D[7];
        dst[0] = s0; dst[1] = s1;
      }
    }
    // Drain completed output row yo = r-15 (received all 16 kernel rows).
    const int yo = r - (KS-1);
    if (yo >= y0) {
      const int slot = yo & 15;
      if (lane < 16) {
        const float inv = 1.0f/256.0f;
        float m0 = fmaxf((accumS[wv][slot][0][lane] + skk[0]) * inv, 1e-30f);
        float m1 = fmaxf((accumS[wv][slot][1][lane] + skk[1]) * inv, 1e-30f);
        float m2 = fmaxf((accumS[wv][slot][2][lane] + skk[2]) * inv, 1e-30f);
        float m3 = fmaxf((accumS[wv][slot][3][lane] + skk[3]) * inv, 1e-30f);
        const float sl = __log2f(m0) + __log2f(m1) + __log2f(m2) + __log2f(m3);
        // mean_c(20*log10(255) - 10*log10(mse)) = 48.1308 - 2.5*log10(2)*sum(log2)
        const float psnr = 48.1308036087f - 0.7525749892f * sl;
        const int xo = x0 + lane;
        if (xo < OUT_W) out[(size_t)yo*OUT_W + xo] = psnr;
        accumS[wv][slot][0][lane] = 0.0f;   // recycle slot for row yo+16
        accumS[wv][slot][1][lane] = 0.0f;
        accumS[wv][slot][2][lane] = 0.0f;
        accumS[wv][slot][3][lane] = 0.0f;
      }
    }
  }
}

extern "C" void kernel_launch(void* const* d_in, const int* in_sizes, int n_in,
                              void* d_out, int out_size, void* d_ws, size_t ws_size,
                              hipStream_t stream) {
  const float* x  = (const float*)d_in[0];   // (1,4,2048,2048) f32
  const float* kn = (const float*)d_in[1];   // (1,4,16,16) f32
  float* out = (float*)d_out;                // (2033,2033) f32
  (void)in_sizes; (void)n_in; (void)out_size; (void)d_ws; (void)ws_size;
  dim3 grid((STRIPS_X * STRIPS_Y) / NWAVES);  // 512 blocks
  dim3 block(BLOCK);                          // 256 threads = 8 wave32
  hipLaunchKernelGGL(psnr_conv_wmma, grid, block, 0, stream, x, kn, out);
}
